// MultiHeadAttention_20461224198727
// MI455X (gfx1250) — matrix operand order repaired and checked by reading, not compiled, not run on hardware
//
#include <hip/hip_runtime.h>

// ---------------------------------------------------------------------------
// MHA forward for MI455X (gfx1250, wave32, bf16 WMMA, async LDS copies)
//   b=2, s=2048, h=1024, heads=16, d=64
// Pipeline:
//   0) cvt_kernel        : f32 -> bf16 for x, Wq, Wk, Wv (one-time, HBM-bound)
//   1) qkv_proj_kernel   : fused Q/K/V GEMM, bf16 WMMA, async double-buffered
//                          tiles; Q scaled by log2(e); V stored transposed
//   2) mha_flash_kernel  : flash attention, 64-wide KV tiles via async copies,
//                          online softmax in exp2 domain, bf16 WMMA
// ---------------------------------------------------------------------------

#define HID   1024
#define SEQ   2048
#define NH    16
#define DH    64
#define BATCH 2
#define LOG2E 1.44269504088896340736f

typedef __attribute__((ext_vector_type(16))) __bf16 v16bf;
typedef __attribute__((ext_vector_type(8)))  float  v8f;

union FragBF {
  v16bf        bf;
  unsigned int w[8];
};

// f32 -> bf16 via native convert (v_cvt_pk_bf16_f32 on gfx1250)
__device__ __forceinline__ unsigned short f2bf(float f) {
  union { __bf16 h; unsigned short u; } c;
  c.h = (__bf16)f;
  return c.u;
}

// LDS byte offset of a __shared__ pointer (flat LDS aperture: low 32 bits)
__device__ __forceinline__ unsigned lds_off(const void* p) {
  return (unsigned)(size_t)p;
}

// Async global->LDS 16B copy (ASYNCcnt-tracked, no VGPR data staging)
__device__ __forceinline__ void async_b128(unsigned lds_byte, const void* gaddr) {
  asm volatile("global_load_async_to_lds_b128 %0, %1, off"
               :: "v"(lds_byte), "v"(gaddr) : "memory");
}
#define WAIT_ASYNC_8() asm volatile("s_wait_asynccnt 0x8" ::: "memory")
#define WAIT_ASYNC_4() asm volatile("s_wait_asynccnt 0x4" ::: "memory")
#define WAIT_ASYNC_0() asm volatile("s_wait_asynccnt 0x0" ::: "memory")
#define WAIT_DS_0()    asm volatile("s_wait_dscnt 0x0"    ::: "memory")

// A-matrix fragment (16x32 bf16). Lane holds row M = lane%16.
// VGPR r (halves 2r,2r+1) holds K = (r&3)*2 + (r>>2)*16 + (lane>=16)*8 , +1
__device__ __forceinline__ v16bf frag_a(const unsigned short* rowPtr, int hi) {
  FragBF f;
#pragma unroll
  for (int r = 0; r < 8; ++r) {
    int k = ((r & 3) << 1) + ((r >> 2) << 4) + (hi << 3);
    f.w[r] = *(const unsigned int*)(rowPtr + k);
  }
  return f.bf;
}

// B-matrix fragment (32x16 bf16). Lane holds column N = lane%16.
// VGPR r (halves 2r,2r+1) holds K = (r&3)*2 + (r>>2)*16 + (lane>=16)*8 , +1
__device__ __forceinline__ v16bf frag_b(const unsigned short* colPtr, int hi) {
  FragBF f;
#pragma unroll
  for (int r = 0; r < 8; ++r) {
    int k = ((r & 3) << 1) + ((r >> 2) << 4) + (hi << 3);
    f.w[r] = *(const unsigned int*)(colPtr + k);
  }
  return f.bf;
}

__device__ __forceinline__ v8f wmma_bf16(v16bf a, v16bf b, v8f c) {
  return __builtin_amdgcn_wmma_f32_16x16x32_bf16(false, a, false, b, (short)0, c,
                                                 false, false);
}

__device__ __forceinline__ float rowmax16(float v) {
  v = fmaxf(v, __shfl_xor(v, 1, 32));
  v = fmaxf(v, __shfl_xor(v, 2, 32));
  v = fmaxf(v, __shfl_xor(v, 4, 32));
  v = fmaxf(v, __shfl_xor(v, 8, 32));
  return v;
}
__device__ __forceinline__ float rowsum16(float v) {
  v += __shfl_xor(v, 1, 32);
  v += __shfl_xor(v, 2, 32);
  v += __shfl_xor(v, 4, 32);
  v += __shfl_xor(v, 8, 32);
  return v;
}

// ---------------------------------------------------------------------------
// Kernel 0: f32 -> bf16 bulk convert (8 elems/thread, vectorized)
// ---------------------------------------------------------------------------
__global__ __launch_bounds__(256) void cvt_kernel(const float* __restrict__ src,
                                                  unsigned short* __restrict__ dst,
                                                  int n) {
  int i = (blockIdx.x * 256 + threadIdx.x) * 8;
  if (i + 8 > n) return;
  float4 a = *(const float4*)(src + i);
  float4 b = *(const float4*)(src + i + 4);
  union { unsigned short s[8]; uint4 v; } p;
  p.s[0] = f2bf(a.x); p.s[1] = f2bf(a.y); p.s[2] = f2bf(a.z); p.s[3] = f2bf(a.w);
  p.s[4] = f2bf(b.x); p.s[5] = f2bf(b.y); p.s[6] = f2bf(b.z); p.s[7] = f2bf(b.w);
  *(uint4*)(dst + i) = p.v;
}

// ---------------------------------------------------------------------------
// Kernel 1: fused QKV projection (bf16 in, bf16 out).
// Grid (64,16), block 128 (4 waves). Each wave: 16x64 slice of Q,K,V.
// Tiles stream via async global->LDS copies, double-buffered.
// Q output pre-scaled by log2(e); V output stored transposed [b,n,d,s].
// ---------------------------------------------------------------------------
__global__ __launch_bounds__(128) void qkv_proj_kernel(
    const unsigned short* __restrict__ xb,
    const unsigned short* __restrict__ wqb,
    const unsigned short* __restrict__ wkb,
    const unsigned short* __restrict__ wvb,
    unsigned short* __restrict__ q_ws,
    unsigned short* __restrict__ k_ws,
    unsigned short* __restrict__ v_ws) {
  __shared__ unsigned short Xs[2][64 * 32];
  __shared__ unsigned short Wt[3][2][64 * 32];

  const int t    = threadIdx.x;
  const int lane = t & 31;
  const int wave = t >> 5;   // 0..3
  const int hi   = lane >> 4;
  const int lo   = lane & 15;

  const int mbase = blockIdx.x * 64;  // rows of x (b*s flattened)
  const int obase = blockIdx.y * 64;  // output features

  const v8f zf = {0.f, 0.f, 0.f, 0.f, 0.f, 0.f, 0.f, 0.f};
  v8f accq[4], acck[4], accv[4];
#pragma unroll
  for (int i = 0; i < 4; ++i) { accq[i] = zf; acck[i] = zf; accv[i] = zf; }

  // issue one 64x32 K-chunk of x + 3 weight tiles into buffer `buf`
  auto issue = [&](int kk, int buf) {
#pragma unroll
    for (int i = 0; i < 2; ++i) {
      int c   = i * 128 + t;    // 0..255
      int row = c >> 2;         // 0..63
      int col = (c & 3) * 8;    // 0,8,16,24
      int dst = row * 32 + col;
      async_b128(lds_off(&Xs[buf][dst]),    xb  + (size_t)(mbase + row) * HID + kk + col);
      async_b128(lds_off(&Wt[0][buf][dst]), wqb + (size_t)(obase + row) * HID + kk + col);
      async_b128(lds_off(&Wt[1][buf][dst]), wkb + (size_t)(obase + row) * HID + kk + col);
      async_b128(lds_off(&Wt[2][buf][dst]), wvb + (size_t)(obase + row) * HID + kk + col);
    }
  };

  issue(0, 0);
  for (int it = 0; it < HID / 32; ++it) {
    const int buf = it & 1;
    if (it + 1 < HID / 32) { issue((it + 1) * 32, buf ^ 1); WAIT_ASYNC_8(); }
    else                   { WAIT_ASYNC_0(); }
    __syncthreads();

    v16bf a = frag_a(&Xs[buf][(wave * 16 + lo) * 32], hi);
#pragma unroll
    for (int tt = 0; tt < 4; ++tt) {
      v16bf bq = frag_b(&Wt[0][buf][(tt * 16 + lo) * 32], hi);
      v16bf bk = frag_b(&Wt[1][buf][(tt * 16 + lo) * 32], hi);
      v16bf bv = frag_b(&Wt[2][buf][(tt * 16 + lo) * 32], hi);
      accq[tt] = wmma_bf16(a, bq, accq[tt]);
      acck[tt] = wmma_bf16(a, bk, acck[tt]);
      accv[tt] = wmma_bf16(a, bv, accv[tt]);
    }
    __syncthreads();
  }

  // Store: Q (log2e-scaled) and K in [b,n,s,d]; V transposed in [b,n,d,s].
#pragma unroll
  for (int tt = 0; tt < 4; ++tt) {
#pragma unroll
    for (int r = 0; r < 8; ++r) {
      int m = mbase + wave * 16 + r + 8 * hi;  // [0,4096)
      int o = obase + tt * 16 + lo;            // [0,1024)
      int b = m >> 11, s = m & (SEQ - 1);
      int n = o >> 6,  d = o & (DH - 1);
      size_t bn  = (size_t)(b * NH + n);
      size_t dqk = (bn * SEQ + s) * DH + d;
      q_ws[dqk] = f2bf(accq[tt][r] * LOG2E);
      k_ws[dqk] = f2bf(acck[tt][r]);
      v_ws[(bn * DH + d) * SEQ + s] = f2bf(accv[tt][r]);
    }
  }
}

// ---------------------------------------------------------------------------
// Kernel 2: flash attention. Grid (2048/128, 32); block 256 (8 waves),
// each wave owns 16 Q rows. KV streamed in 64-wide tiles via async copies,
// double-buffered. Softmax in exp2 domain (Q pre-scaled by log2 e).
// ---------------------------------------------------------------------------
__global__ __launch_bounds__(256) void mha_flash_kernel(
    const unsigned short* __restrict__ q_ws,
    const unsigned short* __restrict__ k_ws,
    const unsigned short* __restrict__ v_ws,   // transposed [b,n,d,s]
    float* __restrict__ out) {
  __shared__ unsigned short Kt[2][64 * 64];   // [kv][d]
  __shared__ unsigned short Vt[2][64 * 64];   // [d][kv]
  __shared__ unsigned short Pt[8][16 * 64];   // per-wave P staging

  const int t    = threadIdx.x;
  const int lane = t & 31;
  const int wave = t >> 5;   // 0..7
  const int hi   = lane >> 4;
  const int lo   = lane & 15;

  const int bn = blockIdx.y;
  const int q0 = blockIdx.x * 128 + wave * 16;

  // Q fragments (16 rows x 64 d), direct global gather (merges to b128 loads)
  const unsigned short* qbase = q_ws + ((size_t)bn * SEQ + q0) * DH;
  v16bf aq0 = frag_a(qbase + (size_t)lo * DH,      hi);
  v16bf aq1 = frag_a(qbase + (size_t)lo * DH + 32, hi);

  const unsigned short* kg = k_ws + (size_t)bn * SEQ * DH;
  const unsigned short* vg = v_ws + (size_t)bn * DH * SEQ;

  const v8f zf = {0.f, 0.f, 0.f, 0.f, 0.f, 0.f, 0.f, 0.f};
  v8f o_acc[4];
#pragma unroll
  for (int i = 0; i < 4; ++i) o_acc[i] = zf;
  float m_r[8], l_r[8];
#pragma unroll
  for (int r = 0; r < 8; ++r) { m_r[r] = -3.0e38f; l_r[r] = 0.f; }

  // async copy one 64-wide KV tile (K: [kv][d], V already transposed [d][kv])
  auto issueKV = [&](int kv, int buf) {
#pragma unroll
    for (int i = 0; i < 2; ++i) {
      int c   = i * 256 + t;   // 0..511
      int row = c >> 3;        // 0..63
      int col = (c & 7) * 8;   // 0..56
      async_b128(lds_off(&Kt[buf][row * 64 + col]), kg + (size_t)(kv + row) * DH + col);
      async_b128(lds_off(&Vt[buf][row * 64 + col]), vg + (size_t)row * SEQ + kv + col);
    }
  };

  const int NT = SEQ / 64;  // 32 tiles
  issueKV(0, 0);
  for (int it = 0; it < NT; ++it) {
    const int buf = it & 1;
    if (it + 1 < NT) { issueKV((it + 1) * 64, buf ^ 1); WAIT_ASYNC_4(); }
    else             { WAIT_ASYNC_0(); }
    __syncthreads();

    // S' = (log2e * Q) (16x64) . K_tile^T (64x64): 4 16-col subtiles
    v8f s[4];
#pragma unroll
    for (int nt = 0; nt < 4; ++nt) {
      v16bf b0 = frag_b(&Kt[buf][(nt * 16 + lo) * 64 + 0],  hi);
      v16bf b1 = frag_b(&Kt[buf][(nt * 16 + lo) * 64 + 32], hi);
      s[nt] = wmma_bf16(aq0, b0, zf);
      s[nt] = wmma_bf16(aq1, b1, s[nt]);
    }

    // Online softmax (base-2). C-frag row M = r + 8*hi lives across lanes.
    unsigned short* Pw = &Pt[wave][0];
#pragma unroll
    for (int r = 0; r < 8; ++r) {
      float tm = fmaxf(fmaxf(s[0][r], s[1][r]), fmaxf(s[2][r], s[3][r]));
      tm = rowmax16(tm);
      float mnew = fmaxf(m_r[r], tm);
      float corr = __builtin_amdgcn_exp2f(m_r[r] - mnew);
      float p0 = __builtin_amdgcn_exp2f(s[0][r] - mnew);
      float p1 = __builtin_amdgcn_exp2f(s[1][r] - mnew);
      float p2 = __builtin_amdgcn_exp2f(s[2][r] - mnew);
      float p3 = __builtin_amdgcn_exp2f(s[3][r] - mnew);
      float psum = rowsum16((p0 + p1) + (p2 + p3));
      l_r[r] = l_r[r] * corr + psum;
      m_r[r] = mnew;
#pragma unroll
      for (int tt = 0; tt < 4; ++tt) o_acc[tt][r] = o_acc[tt][r] * corr;
      int prow = (r + 8 * hi) * 64;
      Pw[prow +  0 + lo] = f2bf(p0);
      Pw[prow + 16 + lo] = f2bf(p1);
      Pw[prow + 32 + lo] = f2bf(p2);
      Pw[prow + 48 + lo] = f2bf(p3);
    }
    WAIT_DS_0();  // P stores -> P fragment loads (same wave)

    // O += P (16x64) . V_tile (64x64)
    v16bf ap0 = frag_a(&Pw[lo * 64],      hi);
    v16bf ap1 = frag_a(&Pw[lo * 64 + 32], hi);
#pragma unroll
    for (int tt = 0; tt < 4; ++tt) {
      v16bf bv0 = frag_b(&Vt[buf][(tt * 16 + lo) * 64 + 0],  hi);
      v16bf bv1 = frag_b(&Vt[buf][(tt * 16 + lo) * 64 + 32], hi);
      o_acc[tt] = wmma_bf16(ap0, bv0, o_acc[tt]);
      o_acc[tt] = wmma_bf16(ap1, bv1, o_acc[tt]);
    }
    __syncthreads();
  }

  // Epilogue: O / l, scatter to out[b, q, n*64+d] (fp32)
  const int b = bn >> 4, n = bn & 15;
#pragma unroll
  for (int r = 0; r < 8; ++r) {
    float inv = 1.0f / l_r[r];
    int q = q0 + r + 8 * hi;
#pragma unroll
    for (int tt = 0; tt < 4; ++tt) {
      int d = tt * 16 + lo;
      out[((size_t)b * SEQ + q) * HID + n * DH + d] = o_acc[tt][r] * inv;
    }
  }
}

// ---------------------------------------------------------------------------
extern "C" void kernel_launch(void* const* d_in, const int* in_sizes, int n_in,
                              void* d_out, int out_size, void* d_ws, size_t ws_size,
                              hipStream_t stream) {
  const float* x  = (const float*)d_in[0];
  const float* Wq = (const float*)d_in[1];
  const float* Wk = (const float*)d_in[2];
  const float* Wv = (const float*)d_in[3];
  float* out = (float*)d_out;

  const size_t xN = (size_t)BATCH * SEQ * HID;   // 4M
  const size_t wN = (size_t)HID * HID;           // 1M
  const size_t pN = (size_t)BATCH * NH * SEQ * DH;  // 4M

  unsigned short* xb  = (unsigned short*)d_ws;
  unsigned short* wqb = xb  + xN;
  unsigned short* wkb = wqb + wN;
  unsigned short* wvb = wkb + wN;
  unsigned short* qw  = wvb + wN;
  unsigned short* kw  = qw  + pN;
  unsigned short* vw  = kw  + pN;   // total 19M ushort = 38 MB

  cvt_kernel<<<(int)(xN / 2048), 256, 0, stream>>>(x,  xb,  (int)xN);
  cvt_kernel<<<(int)(wN / 2048), 256, 0, stream>>>(Wq, wqb, (int)wN);
  cvt_kernel<<<(int)(wN / 2048), 256, 0, stream>>>(Wk, wkb, (int)wN);
  cvt_kernel<<<(int)(wN / 2048), 256, 0, stream>>>(Wv, wvb, (int)wN);

  dim3 gproj(BATCH * SEQ / 64, HID / 64);   // (64,16)
  qkv_proj_kernel<<<gproj, 128, 0, stream>>>(xb, wqb, wkb, wvb, qw, kw, vw);

  dim3 gattn(SEQ / 128, BATCH * NH);        // (16,32)
  mha_flash_kernel<<<gattn, 256, 0, stream>>>(qw, kw, vw, out);
}
